// MiniAF3ScoreModel_26508538151730
// MI455X (gfx1250) — matrix operand order repaired and checked by reading, not compiled, not run on hardware
//
#include <hip/hip_runtime.h>
#include <math.h>

// ---------------------------------------------------------------------------
// MiniAF3 Pairformer score model for gfx1250 (MI455X), wave32 + WMMA bf16.
// Shapes: B=64, N=100, C_S=64, C_Z=32, H=4, NUM_BLOCKS=4.
// WMMA operands staged in LDS in the exact ISA 7.12.2 register layout so each
// operand is one 32-byte LDS vector load per lane. Pure global->LDS stagings
// use GLOBAL_LOAD_ASYNC_TO_LDS_B128 (ASYNCcnt) per ISA 15.18.3.
// ---------------------------------------------------------------------------

typedef __bf16 bf16;
typedef __attribute__((ext_vector_type(16))) __bf16 v16bf;
typedef __attribute__((ext_vector_type(4)))  __bf16 v4bf;
typedef __attribute__((ext_vector_type(8)))  float  v8f;

#define BN   64
#define NN   100
#define CS   64
#define CZ   32
#define NHEAD 4

__device__ __forceinline__ float gelu_exact(float x) {
  return 0.5f * x * (1.0f + erff(x * 0.70710678118654752f));
}

__device__ __forceinline__ void rowToBIJ(int R, int& b, int& i, int& j) {
  b = R / (NN * NN);
  int rem = R - b * (NN * NN);
  i = rem / NN;
  j = rem - i * NN;
}

// --- async global->LDS copy (CDNA5, tracked by ASYNCcnt) -------------------
__device__ __forceinline__ void async_copy16(void* lds, const void* gptr) {
  unsigned laddr = (unsigned)(unsigned long long)lds;     // LDS byte address
  unsigned long long ga = (unsigned long long)gptr;
  asm volatile("global_load_async_to_lds_b128 %0, %1, off"
               :: "v"(laddr), "v"(ga) : "memory");
}
__device__ __forceinline__ void wait_async() {
  asm volatile("s_wait_asynccnt 0x0" ::: "memory");
}

// --- WMMA operand layouts (CDNA5 ISA 7.12.2) -------------------------------
// A (16x32, bf16): lane = row + 16*((k>>3)&1); elem = (k&7) + 8*((k>>4)&1)
// B (32x16, bf16): lane = 16*((k>>3)&1) + n;   elem = (k&7) + 8*((k>>4)&1)
// Both stored as [lane][16] contiguous bf16 -> one 32B vector load per lane.
__device__ __forceinline__ int aswz(int row, int k) {
  return ((row + (((k >> 3) & 1) << 4)) << 4) + ((k & 7) + (((k >> 4) & 1) << 3));
}

__device__ __forceinline__ v16bf ld16(const bf16* base, int lane) {
  return *(const v16bf*)(base + (lane << 4));
}

// Swizzle a row-major f32 weight [KC*32][RN] into KC*NT B tiles of 512 bf16.
// Columns >= RN are zero-padded (used for the 32x4 zb_w tile).
template <int KC, int NT, int RN>
__device__ __forceinline__ void swizzleB(const float* __restrict__ src, bf16* dst,
                                         int tid, int nthreads) {
  constexpr int total = KC * NT * 512;
  for (int t = tid; t < total; t += nthreads) {
    int tile = t >> 9, idx = t & 511;
    int lane = idx >> 4, e = idx & 15;
    int kc = tile / NT, nt = tile - kc * NT;
    int n = nt * 16 + (lane & 15);
    int k = (kc << 5) + ((lane >> 4) << 3) + (e & 7) + ((e >> 3) << 4);
    dst[t] = (n < RN) ? (bf16)src[k * RN + n] : (bf16)0.0f;
  }
}

__device__ __forceinline__ v8f wmma_bf16(v16bf a, v16bf b, v8f c) {
  return __builtin_amdgcn_wmma_f32_16x16x32_bf16(false, a, false, b, (short)0, c,
                                                 false, false);
}

// Pack 4 consecutive-K f32 values (k0 % 4 == 0) into the A-swizzled bf16 tile.
__device__ __forceinline__ void storeA4(bf16* tileA, int row, int k0, float4 v) {
  v4bf pk = {(bf16)v.x, (bf16)v.y, (bf16)v.z, (bf16)v.w};
  *(v4bf*)&tileA[aswz(row, k0)] = pk;   // 4 consecutive elems, 8B aligned
}

// ---------------------------------------------------------------------------
// t embedding: temb[b,c] = gelu(t[b]*w1[c]+b1[c]) @ w2 + b2
// ---------------------------------------------------------------------------
__global__ void k_temb(const float* __restrict__ t, const float* __restrict__ w1,
                       const float* __restrict__ b1, const float* __restrict__ w2,
                       const float* __restrict__ b2, float* __restrict__ temb) {
  int b = blockIdx.x, c = threadIdx.x;  // 64 threads
  __shared__ float h[CS];
  float tv = t[b];
  h[c] = gelu_exact(tv * w1[c] + b1[c]);
  __syncthreads();
  float acc = b2[c];
  for (int k = 0; k < CS; ++k) acc += h[k] * w2[k * CS + c];
  temb[b * CS + c] = acc;
}

// s init: s[b,n,c] = seq_emb[seq] + pos_emb[n] + temb[b]   (float4 lanes)
__global__ void k_init_s(const int* __restrict__ seq, const float* __restrict__ seq_emb,
                         const float* __restrict__ pos_emb, const float* __restrict__ temb,
                         float* __restrict__ s) {
  int g = blockIdx.x * 128 + threadIdx.x;  // B*N*CS/4 = 102400 quads
  int r = g >> 4, c0 = (g & 15) * 4;
  int b = r / NN, n = r - b * NN;
  float4 a = *(const float4*)(seq_emb + seq[r] * CS + c0);
  float4 p = *(const float4*)(pos_emb + n * CS + c0);
  float4 tv = *(const float4*)(temb + b * CS + c0);
  float4 o;
  o.x = a.x + p.x + tv.x; o.y = a.y + p.y + tv.y;
  o.z = a.z + p.z + tv.z; o.w = a.w + p.w + tv.w;
  *(float4*)(s + r * CS + c0) = o;
}

// z init: z[b,i,j,c] = dist_emb[|i-j|] + ip*ipp_w + ipp_b   (float4 lanes)
__global__ void k_init_z(const float* __restrict__ spins, const float* __restrict__ dist_emb,
                         const float* __restrict__ ipp_w, const float* __restrict__ ipp_b,
                         float* __restrict__ z) {
  int g = blockIdx.x * 256 + threadIdx.x;   // B*N*N*8 = 5,120,000 quads
  int c0 = (g & 7) * 4;
  int R = g >> 3;
  int b, i, j;
  rowToBIJ(R, b, i, j);
  int d = i - j; if (d < 0) d = -d;
  const float* si = spins + (b * NN + i) * 3;
  const float* sj = spins + (b * NN + j) * 3;
  float ip = si[0] * sj[0] + si[1] * sj[1] + si[2] * sj[2];
  float4 de = *(const float4*)(dist_emb + d * CZ + c0);
  float4 pw = *(const float4*)(ipp_w + c0);
  float4 pb = *(const float4*)(ipp_b + c0);
  float4 o;
  o.x = de.x + ip * pw.x + pb.x; o.y = de.y + ip * pw.y + pb.y;
  o.z = de.z + ip * pw.z + pb.z; o.w = de.w + ip * pw.w + pb.w;
  *(float4*)(z + R * CZ + c0) = o;
}

// ---------------------------------------------------------------------------
// zi = s @ W_top, zj = s @ W_bot
// ---------------------------------------------------------------------------
__global__ void __launch_bounds__(128) k_zizj(const float* __restrict__ s,
                                              const float* __restrict__ s2z_w,
                                              float* __restrict__ zi,
                                              float* __restrict__ zj) {
  __shared__ __align__(32) bf16 sW[8 * 512];    // KC=4 (Wtop 0..1, Wbot 2..3), NT=2
  __shared__ __align__(32) bf16 sA[4][1024];    // 2 A tiles per wave
  int tid = threadIdx.x;
  swizzleB<4, 2, 32>(s2z_w, sW, tid, 128);
  int w = tid >> 5, lane = tid & 31;
  int tile = blockIdx.x * 4 + w, R0 = tile * 16;
  for (int t4 = lane; t4 < 256; t4 += 32) {
    int e0 = t4 * 4, r = e0 >> 6, c = e0 & 63;
    float4 v4 = *(const float4*)(s + (R0 + r) * CS + c);
    storeA4(&sA[w][(c >> 5) * 512], r, c & 31, v4);
  }
  __syncthreads();
  v16bf a0 = ld16(sA[w], lane);
  v16bf a1 = ld16(sA[w] + 512, lane);
  int col16 = lane & 15, hi = lane >> 4;
#pragma unroll
  for (int o = 0; o < 2; ++o) {
    float* out = o ? zj : zi;
#pragma unroll
    for (int nt = 0; nt < 2; ++nt) {
      v8f acc = {};
      acc = wmma_bf16(a0, ld16(sW + ((o * 2 + 0) * 2 + nt) * 512, lane), acc);
      acc = wmma_bf16(a1, ld16(sW + ((o * 2 + 1) * 2 + nt) * 512, lane), acc);
      int col = nt * 16 + col16;
#pragma unroll
      for (int v = 0; v < 8; ++v)
        out[(R0 + v + 8 * hi) * CZ + col] = acc[v];
    }
  }
}

// ---------------------------------------------------------------------------
// Fused z update per 16-row tile of z:
//   z1 = z + zi[b,i] + zj[b,j] + s2z_b
//   z_new = z1 + W2 @ gelu(W1 @ LN(z1));  bias = z_new @ zb_w + zb_b
// ---------------------------------------------------------------------------
__global__ void __launch_bounds__(128) k_zupdate(
    float* __restrict__ z, const float* __restrict__ zi, const float* __restrict__ zj,
    const float* __restrict__ s2z_b, const float* __restrict__ nz_g,
    const float* __restrict__ nz_b, const float* __restrict__ w1,
    const float* __restrict__ b1, const float* __restrict__ w2,
    const float* __restrict__ b2, const float* __restrict__ zb_w,
    const float* __restrict__ zb_b, float* __restrict__ bias) {
  __shared__ __align__(32) bf16 sW1[4 * 512];   // KC=1, NT=4
  __shared__ __align__(32) bf16 sW2[4 * 512];   // KC=2, NT=2
  __shared__ __align__(32) bf16 sZb[512];       // KC=1, NT=1 (cols 4..15 zero)
  __shared__ float sB1[64], sB2[CZ], sSzb[CZ], sG[CZ], sBt[CZ], sZbB[NHEAD];
  __shared__ __align__(16) float z1t[4][16 * CZ];
  __shared__ __align__(32) bf16 hA[4][512];     // A tile (h, then z_new)
  __shared__ __align__(32) bf16 gA[4][1024];    // 2 A tiles (gelu output)
  int tid = threadIdx.x;
  int w = tid >> 5, lane = tid & 31;
  int tile = blockIdx.x * 4 + w, R0 = tile * 16;
  int col16 = lane & 15, hi = lane >> 4;
  // async-stage this wave's 16x32 z tile into LDS (pure copy, ASYNCcnt)
  for (int t4 = lane; t4 < 128; t4 += 32)
    async_copy16(&z1t[w][t4 * 4], z + R0 * CZ + t4 * 4);
  swizzleB<1, 4, 64>(w1, sW1, tid, 128);
  swizzleB<2, 2, 32>(w2, sW2, tid, 128);
  swizzleB<1, 1, 4>(zb_w, sZb, tid, 128);
  if (tid < 64) sB1[tid] = b1[tid];
  if (tid < CZ) { sB2[tid] = b2[tid]; sSzb[tid] = s2z_b[tid]; sG[tid] = nz_g[tid]; sBt[tid] = nz_b[tid]; }
  if (tid < NHEAD) sZbB[tid] = zb_b[tid];
  wait_async();
  // phase 1: z1 += zi + zj + s2z_b  (in place; same lane<->quad mapping as issue)
  for (int t4 = lane; t4 < 128; t4 += 32) {
    int e0 = t4 * 4, r = e0 >> 5, c = e0 & 31;
    int R = R0 + r, b_, i_, j_;
    rowToBIJ(R, b_, i_, j_);
    float4 zv = *(const float4*)&z1t[w][e0];
    float4 a4 = *(const float4*)(zi + (b_ * NN + i_) * CZ + c);
    float4 b4 = *(const float4*)(zj + (b_ * NN + j_) * CZ + c);
    float4 o;
    o.x = zv.x + a4.x + b4.x + sSzb[c + 0];
    o.y = zv.y + a4.y + b4.y + sSzb[c + 1];
    o.z = zv.z + a4.z + b4.z + sSzb[c + 2];
    o.w = zv.w + a4.w + b4.w + sSzb[c + 3];
    *(float4*)&z1t[w][e0] = o;
  }
  __syncthreads();
  // phase 2: LayerNorm -> hA (bf16, A-swizzled)
  if (lane < 16) {
    int r = lane;
    float s1 = 0.f, s2 = 0.f;
    for (int c4 = 0; c4 < 8; ++c4) {
      float4 x = *(const float4*)&z1t[w][r * CZ + c4 * 4];
      s1 += x.x + x.y + x.z + x.w;
      s2 += x.x * x.x + x.y * x.y + x.z * x.z + x.w * x.w;
    }
    float mu = s1 * (1.0f / CZ), var = s2 * (1.0f / CZ) - mu * mu;
    float rs = rsqrtf(var + 1e-5f);
    for (int c4 = 0; c4 < 8; ++c4) {
      int c = c4 * 4;
      float4 x = *(const float4*)&z1t[w][r * CZ + c];
      float4 o;
      o.x = (x.x - mu) * rs * sG[c + 0] + sBt[c + 0];
      o.y = (x.y - mu) * rs * sG[c + 1] + sBt[c + 1];
      o.z = (x.z - mu) * rs * sG[c + 2] + sBt[c + 2];
      o.w = (x.w - mu) * rs * sG[c + 3] + sBt[c + 3];
      storeA4(hA[w], r, c, o);
    }
  }
  __syncthreads();
  // phase 3: G = gelu(h @ W1 + b1)  -> gA (2 A tiles)
  {
    v16bf a = ld16(hA[w], lane);
#pragma unroll
    for (int nt = 0; nt < 4; ++nt) {
      v8f acc = {};
      acc = wmma_bf16(a, ld16(sW1 + nt * 512, lane), acc);
      int col = nt * 16 + col16;
      bf16* dst = &gA[w][(col >> 5) * 512];
      int kk = col & 31;
#pragma unroll
      for (int v = 0; v < 8; ++v)
        dst[aswz(v + 8 * hi, kk)] = (bf16)gelu_exact(acc[v] + sB1[col]);
    }
  }
  __syncthreads();
  // phase 4: z_new = z1 + G @ W2 + b2 -> overwrite z1t (f32) and hA (bf16)
  {
    v16bf a0 = ld16(gA[w], lane);
    v16bf a1 = ld16(gA[w] + 512, lane);
#pragma unroll
    for (int nt = 0; nt < 2; ++nt) {
      v8f acc = {};
      acc = wmma_bf16(a0, ld16(sW2 + (0 * 2 + nt) * 512, lane), acc);
      acc = wmma_bf16(a1, ld16(sW2 + (1 * 2 + nt) * 512, lane), acc);
      int col = nt * 16 + col16;
#pragma unroll
      for (int v = 0; v < 8; ++v) {
        int r = v + 8 * hi;
        float zn = acc[v] + sB2[col] + z1t[w][r * CZ + col];
        z1t[w][r * CZ + col] = zn;
        hA[w][aswz(r, col)] = (bf16)zn;
      }
    }
  }
  __syncthreads();
  // phase 5a: vectorized z writeback
  for (int t4 = lane; t4 < 128; t4 += 32)
    *(float4*)(z + R0 * CZ + t4 * 4) = *(const float4*)&z1t[w][t4 * 4];
  // phase 5b: bias = z_new @ zb_w + zb_b  (cols 0..3 valid)
  {
    v16bf a = ld16(hA[w], lane);
    v8f acc = {};
    acc = wmma_bf16(a, ld16(sZb, lane), acc);
    if (col16 < NHEAD) {
#pragma unroll
      for (int v = 0; v < 8; ++v) {
        int R = R0 + v + 8 * hi, b_, i_, j_;
        rowToBIJ(R, b_, i_, j_);
        bias[((b_ * NHEAD + col16) * NN + i_) * NN + j_] = acc[v] + sZbB[col16];
      }
    }
  }
}

// ---------------------------------------------------------------------------
// qkv = LN(s) @ qkv_w + qkv_b     [B*N, 192]
// ---------------------------------------------------------------------------
__global__ void __launch_bounds__(128) k_qkv(const float* __restrict__ s,
                                             const float* __restrict__ qkv_w,
                                             const float* __restrict__ qkv_b,
                                             const float* __restrict__ ns_g,
                                             const float* __restrict__ ns_b,
                                             float* __restrict__ qkv) {
  __shared__ __align__(32) bf16 sW[24 * 512];   // KC=2, NT=12
  __shared__ __align__(16) float stF[4][16 * CS];
  __shared__ __align__(32) bf16 sA[4][1024];
  __shared__ float sQB[192], sG[CS], sBt[CS];
  int tid = threadIdx.x;
  int w = tid >> 5, lane = tid & 31;
  int tile = blockIdx.x * 4 + w, R0 = tile * 16;
  // async-stage the s tile (pure copy)
  for (int t4 = lane; t4 < 256; t4 += 32)
    async_copy16(&stF[w][t4 * 4], s + R0 * CS + t4 * 4);
  swizzleB<2, 12, 192>(qkv_w, sW, tid, 128);
  for (int t = tid; t < 192; t += 128) sQB[t] = qkv_b[t];
  if (tid < CS) { sG[tid] = ns_g[tid]; sBt[tid] = ns_b[tid]; }
  wait_async();
  __syncthreads();
  if (lane < 16) {
    int r = lane;
    float s1 = 0.f, s2 = 0.f;
    for (int c4 = 0; c4 < 16; ++c4) {
      float4 x = *(const float4*)&stF[w][r * CS + c4 * 4];
      s1 += x.x + x.y + x.z + x.w;
      s2 += x.x * x.x + x.y * x.y + x.z * x.z + x.w * x.w;
    }
    float mu = s1 * (1.0f / CS), var = s2 * (1.0f / CS) - mu * mu;
    float rs = rsqrtf(var + 1e-5f);
    for (int c4 = 0; c4 < 16; ++c4) {
      int c = c4 * 4;
      float4 x = *(const float4*)&stF[w][r * CS + c];
      float4 o;
      o.x = (x.x - mu) * rs * sG[c + 0] + sBt[c + 0];
      o.y = (x.y - mu) * rs * sG[c + 1] + sBt[c + 1];
      o.z = (x.z - mu) * rs * sG[c + 2] + sBt[c + 2];
      o.w = (x.w - mu) * rs * sG[c + 3] + sBt[c + 3];
      storeA4(&sA[w][(c >> 5) * 512], r, c & 31, o);
    }
  }
  __syncthreads();
  v16bf a0 = ld16(sA[w], lane);
  v16bf a1 = ld16(sA[w] + 512, lane);
  int col16 = lane & 15, hi = lane >> 4;
  for (int nt = 0; nt < 12; ++nt) {
    v8f acc = {};
    acc = wmma_bf16(a0, ld16(sW + (0 * 12 + nt) * 512, lane), acc);
    acc = wmma_bf16(a1, ld16(sW + (1 * 12 + nt) * 512, lane), acc);
    int col = nt * 16 + col16;
#pragma unroll
    for (int v = 0; v < 8; ++v)
      qkv[(R0 + v + 8 * hi) * 192 + col] = acc[v] + sQB[col];
  }
}

// ---------------------------------------------------------------------------
// attention core per (b, h): softmax(q@k^T/4 + bias, mask) @ v
// logits cached in LDS so the second softmax pass does no re-compute.
// ---------------------------------------------------------------------------
__global__ void __launch_bounds__(128) k_attn(const float* __restrict__ qkv,
                                              const float* __restrict__ bias,
                                              const int* __restrict__ mask,
                                              float* __restrict__ aout) {
  int bh = blockIdx.x, b = bh >> 2, h = bh & 3;
  __shared__ __align__(16) float q[NN * 16], k[NN * 16], v[NN * 16];
  __shared__ float lg[NN * NN];   // 40 KB logits cache
  for (int t = threadIdx.x; t < NN * 4; t += blockDim.x) {
    int i = t >> 2, d0 = (t & 3) * 4;
    const float* base = qkv + (b * NN + i) * 192 + h * 16 + d0;
    async_copy16(&q[i * 16 + d0], base);
    async_copy16(&k[i * 16 + d0], base + 64);
    async_copy16(&v[i * 16 + d0], base + 128);
  }
  wait_async();
  __syncthreads();
  int i = threadIdx.x;
  if (i < NN) {
    float qr[16];
#pragma unroll
    for (int d = 0; d < 16; ++d) qr[d] = q[i * 16 + d];
    const float* brow = bias + (bh * NN + i) * NN;
    const int*  mrow = mask + b * NN;
    float* lrow = &lg[i * NN];
    float mx = -3.0e38f;
    for (int j = 0; j < NN; ++j) {
      float dot = 0.f;
#pragma unroll
      for (int d = 0; d < 16; ++d) dot += qr[d] * k[j * 16 + d];
      float l = dot * 0.25f + brow[j];
      l = mrow[j] ? l : -1e9f;
      lrow[j] = l;
      mx = fmaxf(mx, l);
    }
    float sum = 0.f, out[16];
#pragma unroll
    for (int d = 0; d < 16; ++d) out[d] = 0.f;
    for (int j = 0; j < NN; ++j) {
      float e = __expf(lrow[j] - mx);
      sum += e;
#pragma unroll
      for (int d = 0; d < 16; ++d) out[d] += e * v[j * 16 + d];
    }
    float inv = 1.0f / sum;
    float* orow = aout + (b * NN + i) * CS + h * 16;
#pragma unroll
    for (int d = 0; d < 16; ++d) orow[d] = out[d] * inv;
  }
}

// ---------------------------------------------------------------------------
// s += attn_out @ ao_w + ao_b
// ---------------------------------------------------------------------------
__global__ void __launch_bounds__(128) k_ao(const float* __restrict__ aout,
                                            const float* __restrict__ ao_w,
                                            const float* __restrict__ ao_b,
                                            float* __restrict__ s) {
  __shared__ __align__(32) bf16 sW[8 * 512];    // KC=2, NT=4
  __shared__ __align__(32) bf16 sA[4][1024];
  __shared__ float sB[CS];
  int tid = threadIdx.x;
  swizzleB<2, 4, 64>(ao_w, sW, tid, 128);
  if (tid < CS) sB[tid] = ao_b[tid];
  int w = tid >> 5, lane = tid & 31;
  int tile = blockIdx.x * 4 + w, R0 = tile * 16;
  for (int t4 = lane; t4 < 256; t4 += 32) {
    int e0 = t4 * 4, r = e0 >> 6, c = e0 & 63;
    float4 v4 = *(const float4*)(aout + (R0 + r) * CS + c);
    storeA4(&sA[w][(c >> 5) * 512], r, c & 31, v4);
  }
  __syncthreads();
  v16bf a0 = ld16(sA[w], lane);
  v16bf a1 = ld16(sA[w] + 512, lane);
  int col16 = lane & 15, hi = lane >> 4;
#pragma unroll
  for (int nt = 0; nt < 4; ++nt) {
    v8f acc = {};
    acc = wmma_bf16(a0, ld16(sW + (0 * 4 + nt) * 512, lane), acc);
    acc = wmma_bf16(a1, ld16(sW + (1 * 4 + nt) * 512, lane), acc);
    int col = nt * 16 + col16;
#pragma unroll
    for (int v = 0; v < 8; ++v) {
      int idx = (R0 + v + 8 * hi) * CS + col;
      s[idx] = s[idx] + acc[v] + sB[col];
    }
  }
}

// ---------------------------------------------------------------------------
// head: w_pair = gelu(LN(z) @ wh_w1 + b1) @ wh_w2 + b2    (per z row)
// ---------------------------------------------------------------------------
__global__ void __launch_bounds__(128) k_head1(const float* __restrict__ z,
                                               const float* __restrict__ wh_g,
                                               const float* __restrict__ wh_bln,
                                               const float* __restrict__ w1,
                                               const float* __restrict__ b1,
                                               const float* __restrict__ w2,
                                               const float* __restrict__ b2,
                                               float* __restrict__ wp) {
  __shared__ __align__(32) bf16 sW1[2 * 512];   // KC=1, NT=2
  __shared__ float sW2[CZ], sB1[CZ], sG[CZ], sBt[CZ];
  __shared__ __align__(16) float zt[4][16 * CZ];
  __shared__ __align__(32) bf16 hA[4][512];
  int tid = threadIdx.x;
  int w = tid >> 5, lane = tid & 31;
  int tile = blockIdx.x * 4 + w, R0 = tile * 16;
  for (int t4 = lane; t4 < 128; t4 += 32)
    async_copy16(&zt[w][t4 * 4], z + R0 * CZ + t4 * 4);
  swizzleB<1, 2, 32>(w1, sW1, tid, 128);
  if (tid < CZ) { sW2[tid] = w2[tid]; sB1[tid] = b1[tid]; sG[tid] = wh_g[tid]; sBt[tid] = wh_bln[tid]; }
  wait_async();
  __syncthreads();
  if (lane < 16) {
    int r = lane;
    float s1 = 0.f, s2 = 0.f;
    for (int c4 = 0; c4 < 8; ++c4) {
      float4 x = *(const float4*)&zt[w][r * CZ + c4 * 4];
      s1 += x.x + x.y + x.z + x.w;
      s2 += x.x * x.x + x.y * x.y + x.z * x.z + x.w * x.w;
    }
    float mu = s1 * (1.0f / CZ), var = s2 * (1.0f / CZ) - mu * mu;
    float rs = rsqrtf(var + 1e-5f);
    for (int c4 = 0; c4 < 8; ++c4) {
      int c = c4 * 4;
      float4 x = *(const float4*)&zt[w][r * CZ + c];
      float4 o;
      o.x = (x.x - mu) * rs * sG[c + 0] + sBt[c + 0];
      o.y = (x.y - mu) * rs * sG[c + 1] + sBt[c + 1];
      o.z = (x.z - mu) * rs * sG[c + 2] + sBt[c + 2];
      o.w = (x.w - mu) * rs * sG[c + 3] + sBt[c + 3];
      storeA4(hA[w], r, c, o);
    }
  }
  __syncthreads();
  v16bf a = ld16(hA[w], lane);
  int col16 = lane & 15, hi = lane >> 4;
  float part[8];
#pragma unroll
  for (int v = 0; v < 8; ++v) part[v] = 0.f;
#pragma unroll
  for (int nt = 0; nt < 2; ++nt) {
    v8f acc = {};
    acc = wmma_bf16(a, ld16(sW1 + nt * 512, lane), acc);
    int col = nt * 16 + col16;
    float wv = sW2[col];
#pragma unroll
    for (int v = 0; v < 8; ++v)
      part[v] += gelu_exact(acc[v] + sB1[col]) * wv;
  }
  // butterfly-reduce over the 16 columns held in each lane half
#pragma unroll
  for (int off = 1; off < 16; off <<= 1) {
#pragma unroll
    for (int v = 0; v < 8; ++v) part[v] += __shfl_xor(part[v], off, 32);
  }
  if (col16 == 0) {
    float bb = b2[0];
#pragma unroll
    for (int v = 0; v < 8; ++v) wp[R0 + v + 8 * hi] = part[v] + bb;
  }
}

// ---------------------------------------------------------------------------
// final: symmetrize+mask w_pair, raw = w@spins, tangent projection
// ---------------------------------------------------------------------------
__global__ void k_final(const float* __restrict__ wp, const float* __restrict__ spins,
                        const int* __restrict__ mask, float* __restrict__ out) {
  int t = blockIdx.x * blockDim.x + threadIdx.x;  // [0, B*N)
  int b = t / NN, i = t - b * NN;
  float mi = mask[b * NN + i] ? 1.f : 0.f;
  float sx = spins[t * 3 + 0], sy = spins[t * 3 + 1], sz = spins[t * 3 + 2];
  float rx = 0.f, ry = 0.f, rz = 0.f;
  const float* wrow = wp + (b * NN + i) * NN;
  const float* wcol = wp + b * NN * NN + i;
  for (int j = 0; j < NN; ++j) {
    if (j + 8 < NN) __builtin_prefetch(&wcol[(j + 8) * NN], 0, 0);
    float wv = 0.5f * (wrow[j] + wcol[j * NN]);
    float mj = mask[b * NN + j] ? 1.f : 0.f;
    wv = (j == i) ? 0.f : wv * mi * mj;
    rx += wv * spins[(b * NN + j) * 3 + 0];
    ry += wv * spins[(b * NN + j) * 3 + 1];
    rz += wv * spins[(b * NN + j) * 3 + 2];
  }
  float radial = rx * sx + ry * sy + rz * sz;
  out[t * 3 + 0] = (rx - radial * sx) * mi;
  out[t * 3 + 1] = (ry - radial * sy) * mi;
  out[t * 3 + 2] = (rz - radial * sz) * mi;
}

// ---------------------------------------------------------------------------
// launcher
// ---------------------------------------------------------------------------
extern "C" void kernel_launch(void* const* d_in, const int* in_sizes, int n_in,
                              void* d_out, int out_size, void* d_ws, size_t ws_size,
                              hipStream_t stream) {
  (void)in_sizes; (void)n_in; (void)out_size; (void)ws_size;
  const int*   seq      = (const int*)  d_in[0];
  const float* spins    = (const float*)d_in[1];
  const float* tt       = (const float*)d_in[2];
  const int*   mask     = (const int*)  d_in[3];
  const float* seq_emb  = (const float*)d_in[4];
  const float* pos_emb  = (const float*)d_in[5];
  const float* time_w1  = (const float*)d_in[6];
  const float* time_b1  = (const float*)d_in[7];
  const float* time_w2  = (const float*)d_in[8];
  const float* time_b2  = (const float*)d_in[9];
  const float* dist_emb = (const float*)d_in[10];
  const float* ipp_w    = (const float*)d_in[11];
  const float* ipp_b    = (const float*)d_in[12];
  // 4 blocks x 16 params each, dict insertion order:
  // 0 s2z_w 1 s2z_b 2 zmlp_w1 3 zmlp_b1 4 zmlp_w2 5 zmlp_b2 6 qkv_w 7 qkv_b
  // 8 zb_w 9 zb_b 10 ao_w 11 ao_b 12 ns_g 13 ns_b 14 nz_g 15 nz_b
  const float* const* blk = (const float* const*)(d_in + 13);
  int ba = 13 + 4 * 16;
  const float* wh_g   = (const float*)d_in[ba + 0];
  const float* wh_bln = (const float*)d_in[ba + 1];
  const float* wh_w1  = (const float*)d_in[ba + 2];
  const float* wh_b1  = (const float*)d_in[ba + 3];
  const float* wh_w2  = (const float*)d_in[ba + 4];
  const float* wh_b2  = (const float*)d_in[ba + 5];

  float* ws = (float*)d_ws;
  size_t o = 0;
  float* t_emb = ws + o; o += (size_t)BN * CS;              //     4,096
  float* sbuf  = ws + o; o += (size_t)BN * NN * CS;         //   409,600
  float* zbuf  = ws + o; o += (size_t)BN * NN * NN * CZ;    // 20,480,000
  float* zib   = ws + o; o += (size_t)BN * NN * CZ;         //   204,800
  float* zjb   = ws + o; o += (size_t)BN * NN * CZ;         //   204,800
  float* biasb = ws + o; o += (size_t)BN * NHEAD * NN * NN; // 2,560,000
  float* qkvb  = ws + o; o += (size_t)BN * NN * 192;        // 1,228,800
  float* aoutb = ws + o; o += (size_t)BN * NN * CS;         //   409,600
  float* wpb   = ws + o; o += (size_t)BN * NN * NN;         //   640,000

  k_temb  <<<BN, CS, 0, stream>>>(tt, time_w1, time_b1, time_w2, time_b2, t_emb);
  k_init_s<<<(BN * NN * CS / 4) / 128, 128, 0, stream>>>(seq, seq_emb, pos_emb, t_emb, sbuf);
  k_init_z<<<(BN * NN * NN * CZ / 4) / 256, 256, 0, stream>>>(spins, dist_emb, ipp_w, ipp_b, zbuf);

  for (int bi = 0; bi < 4; ++bi) {
    const float* const* p = blk + bi * 16;
    k_zizj   <<<100,   128, 0, stream>>>(sbuf, p[0], zib, zjb);
    k_zupdate<<<10000, 128, 0, stream>>>(zbuf, zib, zjb, p[1], p[14], p[15],
                                         p[2], p[3], p[4], p[5], p[8], p[9], biasb);
    k_qkv    <<<100,   128, 0, stream>>>(sbuf, p[6], p[7], p[12], p[13], qkvb);
    k_attn   <<<BN * NHEAD, 128, 0, stream>>>(qkvb, biasb, mask, aoutb);
    k_ao     <<<100,   128, 0, stream>>>(aoutb, p[10], p[11], sbuf);
  }

  k_head1<<<10000, 128, 0, stream>>>(zbuf, wh_g, wh_bln, wh_w1, wh_b1, wh_w2, wh_b2, wpb);
  k_final<<<(BN * NN) / 128, 128, 0, stream>>>(wpb, spins, mask, (float*)d_out);
}
